// RecurrentRGCN_35871566856825
// MI455X (gfx1250) — compile-verified, matrix operand order checked by reading, not run on hardware
//
#include <hip/hip_runtime.h>

// ---------------- problem constants (match reference) ----------------
#define NN   50000      // entities
#define R2C  480        // relations*2
#define DD   200        // hidden dim
#define TT   4          // timesteps
#define EE   100000     // edges per snapshot
#define LLAY 2          // rgcn layers
#define SLOPE 0.2291666666666667f

// ---------------- padded GEMM dims (guard-free WMMA) ----------------
#define KP1  224        // K=200  -> 7  chunks of 32
#define KP2  416        // K=400  -> 13 chunks of 32
#define NP1  208        // N=200  -> 13 tiles of 16
#define NP2  608        // N=600  -> 38 tiles of 16
#define MPN  50048      // M=50000 -> 391 blocks of 128
#define MPR  512        // M=480   -> 4   blocks of 128

// ---------------- vector types for WMMA ----------------
typedef __attribute__((ext_vector_type(16))) __bf16 v16bf;
typedef __attribute__((ext_vector_type(8)))  __bf16 v8bf;
typedef __attribute__((ext_vector_type(8)))  float  v8f;

static __device__ inline __bf16 f2bf(float f) {
    union { float f; unsigned u; } v; v.f = f;
    unsigned r = v.u + 0x7FFFu + ((v.u >> 16) & 1u);   // round-to-nearest-even
    unsigned short h = (unsigned short)(r >> 16);
    return __builtin_bit_cast(__bf16, h);
}
static __device__ inline float sigmoidf(float x) { return 1.0f / (1.0f + __expf(-x)); }

// =====================================================================
// Guard-free WMMA GEMM.
//   C[Mp,Np] (+)= A[Mp,KP](bf16) * BT[Np,KP](bf16, pre-transposed B)
// All dims padded: Mp %128==0 (8 waves x 16 rows / block), Np %16==0,
// KP %32==0. 8 waves/block share a 16xKP B-tile staged in LDS (row-major
// by n, stride KP+8 to dodge bank conflicts) -> 2x ds_load_b128 per
// fragment. A fragments: 2x global_load_b128, ISA 16-bit lane layout.
// =====================================================================
template <int KP>
__global__ __launch_bounds__(256) void k_wmma_gemm(
    const __bf16* __restrict__ A, const __bf16* __restrict__ BT,
    float* __restrict__ C, int Np, int accum)
{
    constexpr int LS = KP + 8;                 // LDS row stride (elements)
    __shared__ alignas(16) __bf16 Bs[16 * LS];

    const int gn0 = blockIdx.x * 16;
    // cooperative vectorized fill: Bs[n][k] = BT[gn0+n][k]
    for (int idx = threadIdx.x; idx < 16 * (KP / 8); idx += 256) {
        int n  = idx / (KP / 8);
        int k8 = (idx - n * (KP / 8)) * 8;
        *(v8bf*)&Bs[n * LS + k8] =
            *(const v8bf*)&BT[(size_t)(gn0 + n) * KP + k8];
    }
    __syncthreads();

    const int wave = threadIdx.x >> 5;
    const int lane = threadIdx.x & 31;
    const int m0   = (blockIdx.y * 8 + wave) * 16;
    const int half = lane >> 4;                // 0/1: lane K-group select
    const int l15  = lane & 15;
    const int kgrp = half * 8;

    const int   n    = gn0 + l15;
    const int   mrow = m0 + half * 8;

    v8f c;
#pragma unroll
    for (int j = 0; j < 8; ++j) c[j] = 0.0f;
    if (accum) {
#pragma unroll
        for (int v = 0; v < 8; ++v)
            c[v] = C[(size_t)(mrow + v) * Np + n];
    }

    const __bf16* arow = A + (size_t)(m0 + l15) * KP;
    const __bf16* brow = &Bs[l15 * LS];

#pragma unroll
    for (int kk = 0; kk < KP; kk += 32) {
        const v8bf* pa = (const v8bf*)(arow + kk + kgrp);
        v8bf alo = pa[0];                      // K = kk+kgrp   .. +7
        v8bf ahi = pa[2];                      // K = kk+kgrp+16.. +23
        const v8bf* pb = (const v8bf*)(brow + kk + kgrp);
        v8bf blo = pb[0];
        v8bf bhi = pb[2];
        v16bf a = __builtin_shufflevector(alo, ahi,
                    0,1,2,3,4,5,6,7,8,9,10,11,12,13,14,15);
        v16bf b = __builtin_shufflevector(blo, bhi,
                    0,1,2,3,4,5,6,7,8,9,10,11,12,13,14,15);
        c = __builtin_amdgcn_wmma_f32_16x16x32_bf16(
                false, a, false, b, (short)0, c, false, false);
    }

#pragma unroll
    for (int v = 0; v < 8; ++v)
        C[(size_t)(mrow + v) * Np + n] = c[v];
}

// =====================================================================
// elementwise / scatter / reduction kernels
// =====================================================================
__global__ __launch_bounds__(256) void k_l2norm(
    const float* __restrict__ in, float* __restrict__ out, int cols)
{
    __shared__ float red[256];
    const int row = blockIdx.x;
    const int tid = threadIdx.x;
    float v = (tid < cols) ? in[(size_t)row * cols + tid] : 0.0f;
    red[tid] = v * v;
    __syncthreads();
    for (int s = 128; s > 0; s >>= 1) {
        if (tid < s) red[tid] += red[tid + s];
        __syncthreads();
    }
    float scale = 1.0f / fmaxf(sqrtf(red[0]), 1e-12f);
    if (tid < cols) out[(size_t)row * cols + tid] = v * scale;
}

// hh = h;  h_bf = hh_bf = bf16(h)   (bf16 written at padded stride KP1)
__global__ __launch_bounds__(256) void k_prep_step(
    const float* __restrict__ h, float* __restrict__ hh,
    __bf16* __restrict__ h_bf, __bf16* __restrict__ hh_bf)
{
    const int total = NN * DD;
    for (int i = blockIdx.x * 256 + threadIdx.x; i < total; i += gridDim.x * 256) {
        int row = i / DD, d = i - row * DD;
        float v = h[i];
        hh[i] = v;
        __bf16 b = f2bf(v);
        h_bf[(size_t)row * KP1 + d]  = b;
        hh_bf[(size_t)row * KP1 + d] = b;
    }
}

__global__ __launch_bounds__(256) void k_cnt_deg(
    const int* __restrict__ et, const int* __restrict__ dst,
    float* __restrict__ cnt, float* __restrict__ deg, int E)
{
    for (int i = blockIdx.x * 256 + threadIdx.x; i < E; i += gridDim.x * 256) {
        atomicAdd(&cnt[et[i]], 2.0f);      // seg = concat(et, et)
        atomicAdd(&deg[dst[i]], 1.0f);
    }
}

__global__ __launch_bounds__(256) void k_relpool(
    const int* __restrict__ src, const int* __restrict__ dst,
    const int* __restrict__ et, const float* __restrict__ h,
    float* __restrict__ ssum, int E)
{
    const int total = 2 * E * DD;
    for (int i = blockIdx.x * 256 + threadIdx.x; i < total; i += gridDim.x * 256) {
        int e2 = i / DD, d = i - e2 * DD;
        int e  = (e2 < E) ? e2 : (e2 - E);
        int ent = (e2 < E) ? src[e] : dst[e];
        int r   = et[e];
        atomicAdd(&ssum[(size_t)r * DD + d], h[(size_t)ent * DD + d]);
    }
}

// xin_bf[r, 0:200]=emb_rel, [200:400]=mean (padded KP2); h0_bf padded KP1
__global__ __launch_bounds__(256) void k_xinput(
    const float* __restrict__ emb_rel, const float* __restrict__ ssum,
    const float* __restrict__ cnt, const float* __restrict__ h0,
    __bf16* __restrict__ xin_bf, __bf16* __restrict__ h0_bf)
{
    const int total = R2C * DD;
    for (int i = blockIdx.x * 256 + threadIdx.x; i < total; i += gridDim.x * 256) {
        int r = i / DD, d = i - r * DD;
        xin_bf[(size_t)r * KP2 + d]      = f2bf(emb_rel[i]);
        float m = ssum[i] / fmaxf(cnt[r], 1.0f);
        xin_bf[(size_t)r * KP2 + DD + d] = f2bf(m);
        h0_bf[(size_t)r * KP1 + d]       = f2bf(h0[i]);
    }
}

// GRU gates + row l2norm; gi/gh have padded leading dim NP2
__global__ __launch_bounds__(256) void k_gru(
    const float* __restrict__ gi, const float* __restrict__ gh,
    const float* __restrict__ b_ih, const float* __restrict__ b_hh,
    float* __restrict__ h0)
{
    __shared__ float red[256];
    const int r = blockIdx.x;
    const int d = threadIdx.x;
    float hv = 0.0f;
    if (d < DD) {
        float ir  = gi[(size_t)r * NP2 + d]          + b_ih[d];
        float iz  = gi[(size_t)r * NP2 + DD + d]     + b_ih[DD + d];
        float inn = gi[(size_t)r * NP2 + 2 * DD + d] + b_ih[2 * DD + d];
        float hr  = gh[(size_t)r * NP2 + d]          + b_hh[d];
        float hz  = gh[(size_t)r * NP2 + DD + d]     + b_hh[DD + d];
        float hn  = gh[(size_t)r * NP2 + 2 * DD + d] + b_hh[2 * DD + d];
        float rg = sigmoidf(ir + hr);
        float z  = sigmoidf(iz + hz);
        float nn = tanhf(inn + rg * hn);
        float hp = h0[(size_t)r * DD + d];
        hv = (1.0f - z) * nn + z * hp;
    }
    red[d] = hv * hv;
    __syncthreads();
    for (int s = 128; s > 0; s >>= 1) {
        if (d < s) red[d] += red[d + s];
        __syncthreads();
    }
    float scale = 1.0f / fmaxf(sqrtf(red[0]), 1e-12f);
    if (d < DD) h0[(size_t)r * DD + d] = hv * scale;
}

__global__ __launch_bounds__(256) void k_edge_scatter(
    const int* __restrict__ src, const int* __restrict__ dst,
    const int* __restrict__ et, const float* __restrict__ hh,
    const float* __restrict__ h0, float* __restrict__ agg, int E)
{
    const int total = E * DD;
    for (int i = blockIdx.x * 256 + threadIdx.x; i < total; i += gridDim.x * 256) {
        int e = i / DD, d = i - e * DD;
        float v = hh[(size_t)src[e] * DD + d] + h0[(size_t)et[e] * DD + d];
        atomicAdd(&agg[(size_t)dst[e] * DD + d], v);
    }
}

// agg_bf = bf16(agg * 1/max(deg,1))  (padded stride KP1) -> norm folded into A
__global__ __launch_bounds__(256) void k_agg_scale_bf16(
    const float* __restrict__ agg, const float* __restrict__ deg,
    __bf16* __restrict__ agg_bf)
{
    const int total = NN * DD;
    for (int i = blockIdx.x * 256 + threadIdx.x; i < total; i += gridDim.x * 256) {
        int row = i / DD, d = i - row * DD;
        float nrm = 1.0f / fmaxf(deg[row], 1.0f);
        agg_bf[(size_t)row * KP1 + d] = f2bf(agg[i] * nrm);
    }
}

// hh = rrelu(t0);  hh_bf = bf16(hh)   (t0 has ld NP1)
__global__ __launch_bounds__(256) void k_combine(
    const float* __restrict__ t0, float* __restrict__ hh,
    __bf16* __restrict__ hh_bf)
{
    const int total = NN * DD;
    for (int i = blockIdx.x * 256 + threadIdx.x; i < total; i += gridDim.x * 256) {
        int row = i / DD, d = i - row * DD;
        float v = t0[(size_t)row * NP1 + d];
        v = (v >= 0.0f) ? v : v * SLOPE;
        hh[i] = v;
        hh_bf[(size_t)row * KP1 + d] = f2bf(v);
    }
}

__global__ __launch_bounds__(256) void k_final(
    const float* __restrict__ hh, const float* __restrict__ twp,
    const float* __restrict__ tgb, float* __restrict__ h,
    float* __restrict__ hist)
{
    __shared__ float red[256];
    const int row = blockIdx.x;
    const int d = threadIdx.x;
    float hv = (d < DD) ? hh[(size_t)row * DD + d] : 0.0f;
    red[d] = hv * hv;
    __syncthreads();
    for (int s = 128; s > 0; s >>= 1) {
        if (d < s) red[d] += red[d + s];
        __syncthreads();
    }
    float scale = 1.0f / fmaxf(sqrtf(red[0]), 1e-12f);
    if (d < DD) {
        float cur = hv * scale;
        float tw  = sigmoidf(twp[(size_t)row * NP1 + d] + tgb[d]);
        float ho  = h[(size_t)row * DD + d];
        float hn  = tw * cur + (1.0f - tw) * ho;
        h[(size_t)row * DD + d] = hn;
        hist[(size_t)row * DD + d] = hn;
    }
}

// ---- weight prep into padded bf16 buffers (pads pre-zeroed by memset) ----
// dst[r*ldDst + c] = bf16(src[r*C + c])
__global__ __launch_bounds__(256) void k_pad_convert(
    const float* __restrict__ src, __bf16* __restrict__ dst,
    int R, int C, int ldDst)
{
    const int total = R * C;
    for (int i = blockIdx.x * 256 + threadIdx.x; i < total; i += gridDim.x * 256) {
        int r = i / C, c = i - r * C;
        dst[(size_t)r * ldDst + c] = f2bf(src[i]);
    }
}
// dst[c*ldDst + r] = bf16(src[r*C + c])   (transpose: BT[n][k] = B[k][n])
__global__ __launch_bounds__(256) void k_pad_convertT(
    const float* __restrict__ src, __bf16* __restrict__ dst,
    int R, int C, int ldDst)
{
    const int total = R * C;
    for (int i = blockIdx.x * 256 + threadIdx.x; i < total; i += gridDim.x * 256) {
        int r = i / C, c = i - r * C;
        dst[(size_t)c * ldDst + r] = f2bf(src[i]);
    }
}

// =====================================================================
extern "C" void kernel_launch(void* const* d_in, const int* in_sizes, int n_in,
                              void* d_out, int out_size, void* d_ws, size_t ws_size,
                              hipStream_t stream) {
    (void)in_sizes; (void)n_in; (void)out_size; (void)ws_size;
    const int*   p_src  = (const int*)d_in[0];
    const int*   p_dst  = (const int*)d_in[1];
    const int*   p_et   = (const int*)d_in[2];
    const float* p_dyn  = (const float*)d_in[3];
    const float* p_erel = (const float*)d_in[4];
    const float* p_Wih  = (const float*)d_in[5];
    const float* p_Whh  = (const float*)d_in[6];
    const float* p_bih  = (const float*)d_in[7];
    const float* p_bhh  = (const float*)d_in[8];
    const float* p_Wn   = (const float*)d_in[9];
    const float* p_Wl   = (const float*)d_in[10];
    const float* p_Wt   = (const float*)d_in[11];
    const float* p_tgb  = (const float*)d_in[12];
    float* hist = (float*)d_out;

    // ---- carve workspace ----
    char* wp = (char*)d_ws;
    auto carve = [&](size_t bytes) -> void* {
        void* r = (void*)wp;
        wp += (bytes + 255) & ~(size_t)255;
        return r;
    };
    float*  h    = (float*)carve((size_t)NN * DD * 4);
    float*  hh   = (float*)carve((size_t)NN * DD * 4);
    float*  agg  = (float*)carve((size_t)NN * DD * 4);
    float*  t0   = (float*)carve((size_t)MPN * NP1 * 4);
    float*  t2   = (float*)carve((size_t)MPN * NP1 * 4);
    float*  h0   = (float*)carve((size_t)R2C * DD * 4);
    float*  ssum = (float*)carve((size_t)R2C * DD * 4);
    float*  gi   = (float*)carve((size_t)MPR * NP2 * 4);
    float*  gh   = (float*)carve((size_t)MPR * NP2 * 4);
    float*  cnt  = (float*)carve((size_t)R2C * 4);
    float*  deg  = (float*)carve((size_t)NN * 4);

    const size_t szAe = (size_t)MPN * KP1;          // padded entity A operand
    __bf16* h_bf   = (__bf16*)carve(szAe * 2);
    __bf16* hh_bf  = (__bf16*)carve(szAe * 2);
    __bf16* agg_bf = (__bf16*)carve(szAe * 2);
    __bf16* xin_bf = (__bf16*)carve((size_t)MPR * KP2 * 2);
    __bf16* h0_bf  = (__bf16*)carve((size_t)MPR * KP1 * 2);
    __bf16* BTih   = (__bf16*)carve((size_t)NP2 * KP2 * 2);   // = W_ih padded
    __bf16* BThh   = (__bf16*)carve((size_t)NP2 * KP1 * 2);   // = W_hh padded
    __bf16* BTn    = (__bf16*)carve((size_t)LLAY * NP1 * KP1 * 2);
    __bf16* BTl    = (__bf16*)carve((size_t)LLAY * NP1 * KP1 * 2);
    __bf16* BTt    = (__bf16*)carve((size_t)NP1 * KP1 * 2);

    auto cdiv = [](int a, int b) { return (a + b - 1) / b; };
    auto gblocks = [&](int total) { return cdiv(total, 256); };

    // ---- zero padded bf16 operands once (pads stay zero thereafter) ----
    hipMemsetAsync(h_bf,   0, szAe * 2, stream);
    hipMemsetAsync(hh_bf,  0, szAe * 2, stream);
    hipMemsetAsync(agg_bf, 0, szAe * 2, stream);
    hipMemsetAsync(xin_bf, 0, (size_t)MPR * KP2 * 2, stream);
    hipMemsetAsync(h0_bf,  0, (size_t)MPR * KP1 * 2, stream);
    hipMemsetAsync(BTih,   0, (size_t)NP2 * KP2 * 2, stream);
    hipMemsetAsync(BThh,   0, (size_t)NP2 * KP1 * 2, stream);
    hipMemsetAsync(BTn,    0, (size_t)LLAY * NP1 * KP1 * 2, stream);
    hipMemsetAsync(BTl,    0, (size_t)LLAY * NP1 * KP1 * 2, stream);
    hipMemsetAsync(BTt,    0, (size_t)NP1 * KP1 * 2, stream);

    // ---- weight prep ----
    // gi = xin @ W_ih^T : B = W_ih^T, so BT = W_ih (straight, padded)
    k_pad_convert<<<gblocks(3 * DD * 2 * DD), 256, 0, stream>>>(p_Wih, BTih, 3 * DD, 2 * DD, KP2);
    k_pad_convert<<<gblocks(3 * DD * DD),     256, 0, stream>>>(p_Whh, BThh, 3 * DD, DD, KP1);
    // agg @ Wn : B = Wn, so BT = Wn^T
    for (int l = 0; l < LLAY; ++l) {
        k_pad_convertT<<<gblocks(DD * DD), 256, 0, stream>>>(
            p_Wn + (size_t)l * DD * DD, BTn + (size_t)l * NP1 * KP1, DD, DD, KP1);
        k_pad_convertT<<<gblocks(DD * DD), 256, 0, stream>>>(
            p_Wl + (size_t)l * DD * DD, BTl + (size_t)l * NP1 * KP1, DD, DD, KP1);
    }
    k_pad_convertT<<<gblocks(DD * DD), 256, 0, stream>>>(p_Wt, BTt, DD, DD, KP1);

    // ---- init: h = l2norm(dynamic_emb), h0 = emb_rel ----
    k_l2norm<<<NN, 256, 0, stream>>>(p_dyn, h, DD);
    hipMemcpyAsync(h0, p_erel, (size_t)R2C * DD * 4, hipMemcpyDeviceToDevice, stream);

    for (int t = 0; t < TT; ++t) {
        const int* st = p_src + (size_t)t * EE;
        const int* dt = p_dst + (size_t)t * EE;
        const int* et = p_et  + (size_t)t * EE;

        k_prep_step<<<gblocks(NN * DD), 256, 0, stream>>>(h, hh, h_bf, hh_bf);
        hipMemsetAsync(ssum, 0, (size_t)R2C * DD * 4, stream);
        hipMemsetAsync(cnt,  0, (size_t)R2C * 4, stream);
        hipMemsetAsync(deg,  0, (size_t)NN * 4, stream);

        // relation mean pooling + degree
        k_relpool<<<gblocks(2 * EE * DD), 256, 0, stream>>>(st, dt, et, h, ssum, EE);
        k_cnt_deg<<<gblocks(EE), 256, 0, stream>>>(et, dt, cnt, deg, EE);
        k_xinput<<<gblocks(R2C * DD), 256, 0, stream>>>(p_erel, ssum, cnt, h0, xin_bf, h0_bf);

        // GRU on relations (WMMA): gi = xin @ W_ih^T, gh = h0 @ W_hh^T
        {
            dim3 g1(NP2 / 16, MPR / 128);
            k_wmma_gemm<KP2><<<g1, 256, 0, stream>>>(xin_bf, BTih, gi, NP2, 0);
            k_wmma_gemm<KP1><<<g1, 256, 0, stream>>>(h0_bf,  BThh, gh, NP2, 0);
        }
        k_gru<<<R2C, 256, 0, stream>>>(gi, gh, p_bih, p_bhh, h0);

        // time gate pre-activation from start-of-step h (WMMA)
        {
            dim3 g2(NP1 / 16, MPN / 128);
            k_wmma_gemm<KP1><<<g2, 256, 0, stream>>>(h_bf, BTt, t2, NP1, 0);
        }

        // RGCN layers
        for (int l = 0; l < LLAY; ++l) {
            hipMemsetAsync(agg, 0, (size_t)NN * DD * 4, stream);
            k_edge_scatter<<<gblocks(EE * DD), 256, 0, stream>>>(st, dt, et, hh, h0, agg, EE);
            k_agg_scale_bf16<<<gblocks(NN * DD), 256, 0, stream>>>(agg, deg, agg_bf);
            dim3 g2(NP1 / 16, MPN / 128);
            k_wmma_gemm<KP1><<<g2, 256, 0, stream>>>(
                agg_bf, BTn + (size_t)l * NP1 * KP1, t0, NP1, 0);   // (agg*norm)@Wn
            k_wmma_gemm<KP1><<<g2, 256, 0, stream>>>(
                hh_bf,  BTl + (size_t)l * NP1 * KP1, t0, NP1, 1);   // += hh@Wl
            k_combine<<<gblocks(NN * DD), 256, 0, stream>>>(t0, hh, hh_bf);
        }

        // finalize: cur = l2norm(hh); h = tw*cur + (1-tw)*h; emit hist[t]
        k_final<<<NN, 256, 0, stream>>>(hh, t2, p_tgb, h, hist + (size_t)t * NN * DD);
    }
}
